// Model_24481313587903
// MI455X (gfx1250) — matrix operand order repaired and checked by reading, not compiled, not run on hardware
//
#include <hip/hip_runtime.h>
#include <math.h>

// ---------------------------------------------------------------------------
// CDNA5 bf16 WMMA plumbing (gfx1250, wave32)
// ---------------------------------------------------------------------------
typedef __attribute__((ext_vector_type(16))) __bf16 v16bf;
typedef __attribute__((ext_vector_type(8)))  float  v8f;

__device__ __forceinline__ v8f wmma_bf16(v16bf a, v16bf b, v8f c) {
    // (neg_a, A, neg_b, B, c_mod, C, reuse_a, reuse_b)
    return __builtin_amdgcn_wmma_f32_16x16x32_bf16(false, a, false, b, (short)0, c,
                                                   false, false);
}

// A fragment: 16x32 bf16 (MxK), A points at row0, row-major with leading dim lda.
// Lane layout (ISA 7.12.2): lanes 0-15 M=lane, K-halves 0..; lanes 16-31 M=lane-16, K+8.
__device__ __forceinline__ v16bf load_a_frag(const __bf16* A, long lda, int kb, int lane) {
    int m  = lane & 15;
    int kh = (lane >> 4) << 3;                 // +8 for upper half-wave
    const __bf16* row = A + (long)m * lda + kb;
    v16bf a;
#pragma unroll
    for (int v = 0; v < 8; ++v) {
        int k = ((v < 4) ? (2 * v) : (16 + 2 * (v - 4))) + kh;
        a[2 * v]     = row[k];
        a[2 * v + 1] = row[k + 1];
    }
    return a;
}

// B fragment: 32x16 bf16 (KxN). Wt is W transposed: [ncols x K] row-major, so each
// lane reads two runs of 8 contiguous bf16.
// Lanes 0-15: N=lane, K=kb+0..7,16..23 ; lanes 16-31: N=lane-16, K=kb+8..15,24..31.
__device__ __forceinline__ v16bf load_b_frag(const __bf16* Wt, long ldw, int kb, int lane) {
    int n  = lane & 15;
    int kh = (lane >> 4) << 3;                 // +8 for upper half-wave
    const __bf16* row = Wt + (long)n * ldw + kb + kh;
    v16bf b;
#pragma unroll
    for (int e = 0; e < 16; ++e) b[e] = row[(e < 8) ? e : (e + 8)];
    return b;
}

__device__ __forceinline__ float sigf(float x) { return 1.0f / (1.0f + expf(-x)); }

// ---------------------------------------------------------------------------
// Problem constants
// ---------------------------------------------------------------------------
#define NN 256
#define LL 20
#define BB 10
#define DD 128
#define HH 128
#define H2 256
#define KK 16
#define NITEMS 20000

// ---------------------------------------------------------------------------
// Small utility kernels
// ---------------------------------------------------------------------------
__global__ void k_zero(float* p) { p[(long)blockIdx.x * 256 + threadIdx.x] = 0.0f; }

__global__ void k_cvt(const float* __restrict__ src, __bf16* __restrict__ dst) {
    long i = (long)blockIdx.x * 256 + threadIdx.x;
    dst[i] = (__bf16)src[i];
}

// aspProji (ASP,D,H) -> projT (ASP,H,D) bf16   (so H is the "col row" of Wt)
__global__ void k_cvt_projT(const float* __restrict__ proj, __bf16* __restrict__ projT) {
    int i = blockIdx.x * 256 + threadIdx.x;     // 4*128*128 = 65536
    int a = i >> 14, r = i & 16383, h = r >> 7, d = r & 127;
    projT[((long)a * HH + h) * DD + d] = (__bf16)proj[((long)a * DD + d) * HH + h];
}

// ---------------------------------------------------------------------------
// 1) Embedding gather: embs (bf16, NLB x D) + embsB = sum over b (f32, NL x D)
// ---------------------------------------------------------------------------
__global__ void k_gather(const int* __restrict__ seq, const float* __restrict__ itemEmb,
                         __bf16* __restrict__ embsbf, float* __restrict__ embsB) {
    int nl = blockIdx.x;            // N*L = 5120
    int d  = threadIdx.x;           // 128
    float s = 0.0f;
#pragma unroll
    for (int b = 0; b < BB; ++b) {
        int it = seq[nl * BB + b];
        float v = itemEmb[(long)it * DD + d];
        embsbf[((long)nl * BB + b) * DD + d] = (__bf16)v;
        s += v;
    }
    embsB[(long)nl * DD + d] = s;
}

// ---------------------------------------------------------------------------
// 2) u_embs[n] = embs[n].flat dot fc1_w + fc1_b
// ---------------------------------------------------------------------------
__global__ void k_uembs(const int* __restrict__ seq, const float* __restrict__ itemEmb,
                        const float* __restrict__ fc1w, const float* __restrict__ fc1b,
                        float* __restrict__ uembs) {
    __shared__ float red[256];
    int n = blockIdx.x, tid = threadIdx.x;
    float acc = 0.0f;
    for (int i = tid; i < LL * BB * DD; i += 256) {
        int lb = i >> 7, d = i & 127;
        int it = seq[n * (LL * BB) + lb];
        acc += itemEmb[(long)it * DD + d] * fc1w[i];
    }
    red[tid] = acc; __syncthreads();
    for (int s = 128; s > 0; s >>= 1) { if (tid < s) red[tid] += red[tid + s]; __syncthreads(); }
    if (tid == 0) uembs[n] = red[0] + fc1b[0];
}

// ---------------------------------------------------------------------------
// 3) uA[a,n,h] = tanh(tanh(u*proju)), sq[a,n] = ||uA||^2
// ---------------------------------------------------------------------------
__global__ void k_uasq(const float* __restrict__ uembs, const float* __restrict__ proju,
                       float* __restrict__ uA, float* __restrict__ sq) {
    __shared__ float red[128];
    int an = blockIdx.x;            // ASP*N = 1024
    int a = an >> 8, n = an & 255;
    int h = threadIdx.x;            // 128
    float t = tanhf(tanhf(uembs[n] * proju[a * HH + h]));
    uA[(long)an * HH + h] = t;
    red[h] = t * t; __syncthreads();
    for (int s = 64; s > 0; s >>= 1) { if (h < s) red[h] += red[h + s]; __syncthreads(); }
    if (h == 0) sq[an] = red[0];
}

// ---------------------------------------------------------------------------
// 4) per (a,n): 15 nearest neighbors (matches top_k(-dist)[1:], low-index ties)
// ---------------------------------------------------------------------------
__global__ void k_topk(const float* __restrict__ uA, const float* __restrict__ sq,
                       int* __restrict__ nbidx) {
    __shared__ float urow[128];
    __shared__ float vals[256];
    __shared__ float sv[256];
    __shared__ int   si[256];
    int an = blockIdx.x;            // ASP*N
    int a = an >> 8, n = an & 255;
    int m = threadIdx.x;            // 256
    if (m < 128) urow[m] = uA[(long)an * HH + m];
    __syncthreads();
    const float* vm = uA + ((long)a * NN + m) * HH;
    float dot = 0.0f;
    for (int h = 0; h < HH; ++h) dot += urow[h] * vm[h];
    float negd = -(sq[an] + sq[a * NN + m] - 2.0f * dot);
    if (m == n) negd = 1.0f;        // diag dist = -1  -> -dist = 1 (selected first, dropped)
    vals[m] = negd;
    __syncthreads();
    for (int r = 0; r < KK; ++r) {
        sv[m] = vals[m]; si[m] = m; __syncthreads();
        for (int s = 128; s > 0; s >>= 1) {
            if (m < s) {
                float v2 = sv[m + s]; int i2 = si[m + s];
                if (v2 > sv[m] || (v2 == sv[m] && i2 < si[m])) { sv[m] = v2; si[m] = i2; }
            }
            __syncthreads();
        }
        int w = si[0];
        if (m == 0 && r > 0) nbidx[(long)an * (KK - 1) + (r - 1)] = w;
        if (m == w) vals[m] = -3.4e38f;
        __syncthreads();
    }
}

// ---------------------------------------------------------------------------
// 5) sparse neighbor aggregation (sum over b already folded into embsB)
// ---------------------------------------------------------------------------
__global__ void k_nbsum(const int* __restrict__ nbidx, const float* __restrict__ embsB,
                        __bf16* __restrict__ nbsumbf) {
    int apl = blockIdx.x;           // ASP*N*L = 20480
    int d = threadIdx.x;            // 128
    int l = apl % LL, ap = apl / LL;
    const int* ip = nbidx + (long)ap * (KK - 1);
    float acc = 0.0f;
#pragma unroll
    for (int j = 0; j < KK - 1; ++j) {
        int q = ip[j];
        acc += embsB[((long)q * LL + l) * DD + d];
    }
    nbsumbf[(long)apl * DD + d] = (__bf16)acc;
}

// ---------------------------------------------------------------------------
// 6) embs_UI: sum_b tanh(tanh(embs[:,b,:] @ aspProji[a]))  -> neighb[:, :H]
//    (N*L/16=320 Mtiles) x (8 Htiles) x ASP = 10240 tiles, 8 waves/block
// ---------------------------------------------------------------------------
__global__ void k_gemm_ui(const __bf16* __restrict__ embs, const __bf16* __restrict__ projT,
                          float* __restrict__ neighb) {
    int wid = threadIdx.x >> 5, lane = threadIdx.x & 31;
    int tile = blockIdx.x * 8 + wid;
    int ht = tile & 7, mt = (tile >> 3) % 320, a = (tile >> 3) / 320;
    int nl0 = mt * 16;
    const __bf16* W = projT + ((long)a * HH + ht * 16) * DD;
    v8f acc = {};
    for (int b = 0; b < BB; ++b) {
        const __bf16* A = embs + (long)nl0 * (BB * DD) + (long)b * DD;
        v8f c = {};
#pragma unroll
        for (int ks = 0; ks < 4; ++ks)
            c = wmma_bf16(load_a_frag(A, BB * DD, ks * 32, lane),
                          load_b_frag(W, DD, ks * 32, lane), c);
#pragma unroll
        for (int r = 0; r < 8; ++r) acc[r] += tanhf(tanhf(c[r]));
    }
    int col = ht * 16 + (lane & 15), mo = (lane >> 4) * 8;
#pragma unroll
    for (int r = 0; r < 8; ++r) {
        int nl = nl0 + mo + r;
        int n = nl / LL, l = nl % LL;
        neighb[(long)n * (4 * LL * H2) + ((long)a * LL + l) * H2 + col] = acc[r];
    }
}

// ---------------------------------------------------------------------------
// 7) embs_PI: nb_sumB @ dh_w.T + B*dh_b  -> neighb[:, H:]
// ---------------------------------------------------------------------------
__global__ void k_gemm_pi(const __bf16* __restrict__ nbsum, const __bf16* __restrict__ dhwT,
                          const float* __restrict__ dh_b, float* __restrict__ neighb) {
    int wid = threadIdx.x >> 5, lane = threadIdx.x & 31;
    int tile = blockIdx.x * 8 + wid;    // 1280 Mtiles x 8 Htiles
    int ht = tile & 7, mt = tile >> 3;
    int row0 = mt * 16;
    const __bf16* A = nbsum + (long)row0 * DD;
    const __bf16* W = dhwT + (long)(ht * 16) * DD;
    v8f c = {};
#pragma unroll
    for (int ks = 0; ks < 4; ++ks)
        c = wmma_bf16(load_a_frag(A, DD, ks * 32, lane),
                      load_b_frag(W, DD, ks * 32, lane), c);
    int col = ht * 16 + (lane & 15), mo = (lane >> 4) * 8;
    float db = (float)BB * dh_b[col];
#pragma unroll
    for (int r = 0; r < 8; ++r) {
        int row = row0 + mo + r;        // (a*N + p)*L + l
        int a = row / (NN * LL), p = (row / LL) % NN, l = row % LL;
        neighb[(long)p * (4 * LL * H2) + ((long)a * LL + l) * H2 + HH + col] = c[r] + db;
    }
}

// ---------------------------------------------------------------------------
// 8) flat reshape (N,ASP,L,H2)->(N,L,H2,ASP) + max over last: 4 contiguous elems
// ---------------------------------------------------------------------------
__global__ void k_xmax(const float* __restrict__ neighb, __bf16* __restrict__ xbf) {
    long i = (long)blockIdx.x * 256 + threadIdx.x;   // N*L*H2 = 1,310,720
    long n = i / (LL * H2), o = i % (LL * H2);
    const float* p = neighb + n * (4 * LL * H2) + o * 4;
    float mx = fmaxf(fmaxf(p[0], p[1]), fmaxf(p[2], p[3]));
    xbf[i] = (__bf16)mx;
}

// ---------------------------------------------------------------------------
// 9) input projection for a whole LSTM layer: xw[M x 1024] = A[M x 256] @ wih.T
// ---------------------------------------------------------------------------
__global__ void k_gemm_ih(const __bf16* __restrict__ Abf, const __bf16* __restrict__ wihT,
                          float* __restrict__ xw) {
    int wid = threadIdx.x >> 5, lane = threadIdx.x & 31;
    int tile = blockIdx.x * 8 + wid;    // 320 Mtiles x 64 coltiles
    int ct = tile & 63, mt = tile >> 6;
    int row0 = mt * 16, col0 = ct * 16;
    const __bf16* A = Abf + (long)row0 * H2;
    const __bf16* W = wihT + (long)col0 * H2;
    v8f c = {};
#pragma unroll
    for (int ks = 0; ks < 8; ++ks)
        c = wmma_bf16(load_a_frag(A, H2, ks * 32, lane),
                      load_b_frag(W, H2, ks * 32, lane), c);
    int n16 = lane & 15, mo = (lane >> 4) * 8;
#pragma unroll
    for (int r = 0; r < 8; ++r)
        xw[(long)(row0 + mo + r) * (4 * H2) + col0 + n16] = c[r];
}

// ---------------------------------------------------------------------------
// 10) one LSTM time step: g = xw[:,t,:] + h@whh.T + bih + bhh; gate math fused.
//     Each wave computes a 16x16 slice of all 4 gate regions so the c/h update
//     is local.  h double-buffered bf16 across launches.
// ---------------------------------------------------------------------------
__global__ void k_lstm_step(const float* __restrict__ xw, const __bf16* __restrict__ hin,
                            const __bf16* __restrict__ whhT, const float* __restrict__ bih,
                            const float* __restrict__ bhh, float* __restrict__ c_st,
                            __bf16* __restrict__ hout, __bf16* __restrict__ ybf,
                            float* __restrict__ hsum, int t) {
    int wid = threadIdx.x >> 5, lane = threadIdx.x & 31;
    int tile = blockIdx.x * 8 + wid;    // 16 rowtiles x 16 coltiles = 256
    int ct = tile & 15, rt = tile >> 4;
    int row0 = rt * 16, j0 = ct * 16;
    v8f g0 = {}, g1 = {}, g2 = {}, g3 = {};
#pragma unroll
    for (int ks = 0; ks < 8; ++ks) {
        v16bf af = load_a_frag(hin + (long)row0 * H2, H2, ks * 32, lane);
        g0 = wmma_bf16(af, load_b_frag(whhT + (long)(0 * H2 + j0) * H2, H2, ks * 32, lane), g0);
        g1 = wmma_bf16(af, load_b_frag(whhT + (long)(1 * H2 + j0) * H2, H2, ks * 32, lane), g1);
        g2 = wmma_bf16(af, load_b_frag(whhT + (long)(2 * H2 + j0) * H2, H2, ks * 32, lane), g2);
        g3 = wmma_bf16(af, load_b_frag(whhT + (long)(3 * H2 + j0) * H2, H2, ks * 32, lane), g3);
    }
    int j = j0 + (lane & 15), mo = (lane >> 4) * 8;
    float bi0 = bih[j] + bhh[j];
    float bi1 = bih[H2 + j] + bhh[H2 + j];
    float bi2 = bih[2 * H2 + j] + bhh[2 * H2 + j];
    float bi3 = bih[3 * H2 + j] + bhh[3 * H2 + j];
#pragma unroll
    for (int r = 0; r < 8; ++r) {
        int n = row0 + mo + r;
        long xb = ((long)n * LL + t) * (4 * H2);
        float gi = g0[r] + xw[xb + j] + bi0;
        float gf = g1[r] + xw[xb + H2 + j] + bi1;
        float gg = g2[r] + xw[xb + 2 * H2 + j] + bi2;
        float go = g3[r] + xw[xb + 3 * H2 + j] + bi3;
        long hix = (long)n * H2 + j;
        float cn = sigf(gf) * c_st[hix] + sigf(gi) * tanhf(gg);
        float hn = sigf(go) * tanhf(cn);
        c_st[hix] = cn;
        hout[hix] = (__bf16)hn;
        if (ybf)  ybf[((long)n * LL + t) * H2 + j] = (__bf16)hn;
        if (hsum) hsum[hix] += hn;
    }
}

// ---------------------------------------------------------------------------
// 11) h_agg = tanh(hsum) (in place) + bf16 copy
// ---------------------------------------------------------------------------
__global__ void k_hagg(float* __restrict__ hsum, __bf16* __restrict__ haggbf) {
    long i = (long)blockIdx.x * 256 + threadIdx.x;   // 65536
    float v = tanhf(hsum[i]);
    hsum[i] = v;
    haggbf[i] = (__bf16)v;
}

// ---------------------------------------------------------------------------
// 12) logits = h_agg @ out_w.T + out_b  -> d_out  (16 x 1250 tiles)
// ---------------------------------------------------------------------------
__global__ void k_gemm_logits(const __bf16* __restrict__ hagg, const __bf16* __restrict__ outw,
                              const float* __restrict__ outb, float* __restrict__ out) {
    int wid = threadIdx.x >> 5, lane = threadIdx.x & 31;
    int tile = blockIdx.x * 8 + wid;    // 20000
    int ct = tile % 1250, mt = tile / 1250;
    int row0 = mt * 16, col0 = ct * 16;
    const __bf16* A = hagg + (long)row0 * H2;
    const __bf16* W = outw + (long)col0 * H2;
    v8f c = {};
#pragma unroll
    for (int ks = 0; ks < 8; ++ks)
        c = wmma_bf16(load_a_frag(A, H2, ks * 32, lane),
                      load_b_frag(W, H2, ks * 32, lane), c);
    int n16 = lane & 15, mo = (lane >> 4) * 8;
    float bb = outb[col0 + n16];
#pragma unroll
    for (int r = 0; r < 8; ++r)
        out[(long)(row0 + mo + r) * NITEMS + col0 + n16] = c[r] + bb;
}

// ---------------------------------------------------------------------------
// 13) softmax over 20000, in place
// ---------------------------------------------------------------------------
__global__ void k_softmax(float* __restrict__ out) {
    __shared__ float red[256];
    int row = blockIdx.x, tid = threadIdx.x;
    float* p = out + (long)row * NITEMS;
    float mx = -3.4e38f;
    for (int i = tid; i < NITEMS; i += 256) mx = fmaxf(mx, p[i]);
    red[tid] = mx; __syncthreads();
    for (int s = 128; s > 0; s >>= 1) { if (tid < s) red[tid] = fmaxf(red[tid], red[tid + s]); __syncthreads(); }
    float M = red[0]; __syncthreads();
    float sm = 0.0f;
    for (int i = tid; i < NITEMS; i += 256) sm += expf(p[i] - M);
    red[tid] = sm; __syncthreads();
    for (int s = 128; s > 0; s >>= 1) { if (tid < s) red[tid] += red[tid + s]; __syncthreads(); }
    float inv = 1.0f / red[0];
    for (int i = tid; i < NITEMS; i += 256) p[i] = expf(p[i] - M) * inv;
}

// ---------------------------------------------------------------------------
// 14) uHis_h = uHis @ his_w.T + his_b   (K = 20000, 16x16 tiles)
// ---------------------------------------------------------------------------
__global__ void k_gemm_uhis(const __bf16* __restrict__ uhisbf, const __bf16* __restrict__ hiswbf,
                            const float* __restrict__ hisb, float* __restrict__ uhish) {
    int wid = threadIdx.x >> 5, lane = threadIdx.x & 31;
    int tile = blockIdx.x * 8 + wid;    // 256
    int ct = tile & 15, mt = tile >> 4;
    int row0 = mt * 16, col0 = ct * 16;
    const __bf16* A = uhisbf + (long)row0 * NITEMS;
    const __bf16* W = hiswbf + (long)col0 * NITEMS;
    v8f c = {};
    for (int ks = 0; ks < NITEMS / 32; ++ks)
        c = wmma_bf16(load_a_frag(A, NITEMS, ks * 32, lane),
                      load_b_frag(W, NITEMS, ks * 32, lane), c);
    int n16 = lane & 15, mo = (lane >> 4) * 8;
    float bb = hisb[col0 + n16];
#pragma unroll
    for (int r = 0; r < 8; ++r)
        uhish[(long)(row0 + mo + r) * H2 + col0 + n16] = c[r] + bb;
}

// ---------------------------------------------------------------------------
// 15) gate[n] = sigmoid(uHis_h.g1 + g1_b + h_agg.g2 + g2_b)
// ---------------------------------------------------------------------------
__global__ void k_gate(const float* __restrict__ uhish, const float* __restrict__ hagg,
                       const float* __restrict__ g1w, const float* __restrict__ g1b,
                       const float* __restrict__ g2w, const float* __restrict__ g2b,
                       float* __restrict__ gate) {
    __shared__ float red[256];
    int n = blockIdx.x, tid = threadIdx.x;
    red[tid] = uhish[(long)n * H2 + tid] * g1w[tid] + hagg[(long)n * H2 + tid] * g2w[tid];
    __syncthreads();
    for (int s = 128; s > 0; s >>= 1) { if (tid < s) red[tid] += red[tid + s]; __syncthreads(); }
    if (tid == 0) gate[n] = sigf(red[0] + g1b[0] + g2b[0]);
}

// ---------------------------------------------------------------------------
// 16) scores = gate*probs + (1-gate)*uHis (in place on d_out)
// ---------------------------------------------------------------------------
__global__ void k_combine(const float* __restrict__ gate, const float* __restrict__ uHis,
                          float* __restrict__ out) {
    long i = (long)blockIdx.x * 256 + threadIdx.x;   // 5,120,000
    int n = (int)(i / NITEMS);
    float g = gate[n];
    out[i] = g * out[i] + (1.0f - g) * uHis[i];
}

// ---------------------------------------------------------------------------
// Host side
// ---------------------------------------------------------------------------
static char* carve(char*& p, size_t bytes) {
    char* r = p;
    p += (bytes + 255) & ~(size_t)255;
    return r;
}

extern "C" void kernel_launch(void* const* d_in, const int* in_sizes, int n_in,
                              void* d_out, int out_size, void* d_ws, size_t ws_size,
                              hipStream_t stream) {
    (void)in_sizes; (void)n_in; (void)out_size; (void)ws_size;
    const int*   seq     = (const int*)d_in[0];
    const float* uHis    = (const float*)d_in[1];
    const float* itemEmb = (const float*)d_in[2];
    const float* fc1_w   = (const float*)d_in[3];
    const float* fc1_b   = (const float*)d_in[4];
    const float* aspProju= (const float*)d_in[5];
    const float* aspProji= (const float*)d_in[6];
    const float* dh_w    = (const float*)d_in[7];
    const float* dh_b    = (const float*)d_in[8];
    const float* wih0    = (const float*)d_in[9];
    const float* whh0    = (const float*)d_in[10];
    const float* bih0    = (const float*)d_in[11];
    const float* bhh0    = (const float*)d_in[12];
    const float* wih1    = (const float*)d_in[13];
    const float* whh1    = (const float*)d_in[14];
    const float* bih1    = (const float*)d_in[15];
    const float* bhh1    = (const float*)d_in[16];
    const float* out_w   = (const float*)d_in[17];
    const float* out_b   = (const float*)d_in[18];
    const float* his_w   = (const float*)d_in[19];
    const float* his_b   = (const float*)d_in[20];
    const float* g1_w    = (const float*)d_in[21];
    const float* g1_b    = (const float*)d_in[22];
    const float* g2_w    = (const float*)d_in[23];
    const float* g2_b    = (const float*)d_in[24];
    float* out = (float*)d_out;

    char* p = (char*)d_ws;
    __bf16* embsbf  = (__bf16*)carve(p, (size_t)NN * LL * BB * DD * 2);     // 13.1 MB
    float*  embsB   = (float*) carve(p, (size_t)NN * LL * DD * 4);          // 2.6 MB
    float*  uembs   = (float*) carve(p, NN * 4);
    float*  uA      = (float*) carve(p, (size_t)4 * NN * HH * 4);
    float*  sq      = (float*) carve(p, 4 * NN * 4);
    int*    nbidx   = (int*)   carve(p, (size_t)4 * NN * (KK - 1) * 4);
    __bf16* nbsumbf = (__bf16*)carve(p, (size_t)4 * NN * LL * DD * 2);      // 5.2 MB
    float*  neighb  = (float*) carve(p, (size_t)NN * 4 * LL * H2 * 4);      // 21 MB
    __bf16* xbf     = (__bf16*)carve(p, (size_t)NN * LL * H2 * 2);          // 2.6 MB
    float*  xw      = (float*) carve(p, (size_t)NN * LL * 4 * H2 * 4);      // 21 MB
    __bf16* y0bf    = (__bf16*)carve(p, (size_t)NN * LL * H2 * 2);          // 2.6 MB
    __bf16* hA      = (__bf16*)carve(p, (size_t)NN * H2 * 2);
    __bf16* hB      = (__bf16*)carve(p, (size_t)NN * H2 * 2);
    float*  cst     = (float*) carve(p, (size_t)NN * H2 * 4);
    float*  hsum    = (float*) carve(p, (size_t)NN * H2 * 4);
    __bf16* haggbf  = (__bf16*)carve(p, (size_t)NN * H2 * 2);
    float*  uhish   = (float*) carve(p, (size_t)NN * H2 * 4);
    float*  gate    = (float*) carve(p, NN * 4);
    __bf16* projT   = (__bf16*)carve(p, (size_t)4 * HH * DD * 2);
    __bf16* dhwbf   = (__bf16*)carve(p, (size_t)HH * DD * 2);
    __bf16* wih0bf  = (__bf16*)carve(p, (size_t)4 * H2 * H2 * 2);
    __bf16* whh0bf  = (__bf16*)carve(p, (size_t)4 * H2 * H2 * 2);
    __bf16* wih1bf  = (__bf16*)carve(p, (size_t)4 * H2 * H2 * 2);
    __bf16* whh1bf  = (__bf16*)carve(p, (size_t)4 * H2 * H2 * 2);
    __bf16* outwbf  = (__bf16*)carve(p, (size_t)NITEMS * H2 * 2);           // 10.2 MB
    __bf16* hiswbf  = (__bf16*)carve(p, (size_t)H2 * NITEMS * 2);           // 10.2 MB
    __bf16* uhisbf  = (__bf16*)carve(p, (size_t)NN * NITEMS * 2);           // 10.2 MB

    // --- weight / input conversions to bf16 (Wt layouts) ---
    k_cvt_projT<<<256, 256, 0, stream>>>(aspProji, projT);
    k_cvt<<<HH * DD / 256, 256, 0, stream>>>(dh_w, dhwbf);          // dh_w is (H,D) = Wt
    k_cvt<<<4 * H2 * H2 / 256, 256, 0, stream>>>(wih0, wih0bf);
    k_cvt<<<4 * H2 * H2 / 256, 256, 0, stream>>>(whh0, whh0bf);
    k_cvt<<<4 * H2 * H2 / 256, 256, 0, stream>>>(wih1, wih1bf);
    k_cvt<<<4 * H2 * H2 / 256, 256, 0, stream>>>(whh1, whh1bf);
    k_cvt<<<NITEMS * H2 / 256, 256, 0, stream>>>(out_w, outwbf);    // (NITEMS,H2) = Wt
    k_cvt<<<H2 * NITEMS / 256, 256, 0, stream>>>(his_w, hiswbf);    // (H2,NITEMS) = Wt
    k_cvt<<<NN * NITEMS / 256, 256, 0, stream>>>(uHis, uhisbf);

    // --- embedding gather + user embedding ---
    k_gather<<<NN * LL, 128, 0, stream>>>(seq, itemEmb, embsbf, embsB);
    k_uembs<<<NN, 256, 0, stream>>>(seq, itemEmb, fc1_w, fc1_b, uembs);
    k_uasq<<<4 * NN, 128, 0, stream>>>(uembs, aspProju, uA, sq);
    k_topk<<<4 * NN, 256, 0, stream>>>(uA, sq, nbidx);
    k_nbsum<<<4 * NN * LL, 128, 0, stream>>>(nbidx, embsB, nbsumbf);

    // --- aspect GEMMs -> neighb, then max-reshape -> x ---
    k_gemm_ui<<<1280, 256, 0, stream>>>(embsbf, projT, neighb);
    k_gemm_pi<<<1280, 256, 0, stream>>>(nbsumbf, dhwbf, dh_b, neighb);
    k_xmax<<<NN * LL * H2 / 256, 256, 0, stream>>>(neighb, xbf);

    // --- LSTM layer 0 ---
    k_gemm_ih<<<2560, 256, 0, stream>>>(xbf, wih0bf, xw);
    k_zero<<<NN * H2 * 2 / 4 / 256, 256, 0, stream>>>((float*)hA);  // bf16 zeros
    k_zero<<<NN * H2 / 256, 256, 0, stream>>>(cst);
    for (int t = 0; t < LL; ++t) {
        const __bf16* hin = (t & 1) ? hB : hA;
        __bf16* hout = (t & 1) ? hA : hB;
        k_lstm_step<<<32, 256, 0, stream>>>(xw, hin, whh0bf, bih0, bhh0, cst,
                                            hout, y0bf, nullptr, t);
    }

    // --- LSTM layer 1 ---
    k_gemm_ih<<<2560, 256, 0, stream>>>(y0bf, wih1bf, xw);
    k_zero<<<NN * H2 * 2 / 4 / 256, 256, 0, stream>>>((float*)hA);
    k_zero<<<NN * H2 / 256, 256, 0, stream>>>(cst);
    k_zero<<<NN * H2 / 256, 256, 0, stream>>>(hsum);
    for (int t = 0; t < LL; ++t) {
        const __bf16* hin = (t & 1) ? hB : hA;
        __bf16* hout = (t & 1) ? hA : hB;
        k_lstm_step<<<32, 256, 0, stream>>>(xw, hin, whh1bf, bih1, bhh1, cst,
                                            hout, nullptr, hsum, t);
    }
    k_hagg<<<NN * H2 / 256, 256, 0, stream>>>(hsum, haggbf);

    // --- output head ---
    k_gemm_logits<<<2500, 256, 0, stream>>>(haggbf, outwbf, out_b, out);
    k_softmax<<<NN, 256, 0, stream>>>(out);
    k_gemm_uhis<<<32, 256, 0, stream>>>(uhisbf, hiswbf, his_b, uhish);
    k_gate<<<NN, 256, 0, stream>>>(uhish, hsum, g1_w, g1_b, g2_w, g2_b, gate);
    k_combine<<<NN * NITEMS / 256, 256, 0, stream>>>(gate, uHis, out);
}
